// SlidingWindowAttention_87548613362350
// MI455X (gfx1250) — compile-verified, matrix operand order checked by reading, not run on hardware
//
#include <hip/hip_runtime.h>

// ---------------------------------------------------------------------------
// Problem constants (from reference)
// ---------------------------------------------------------------------------
#define BATCH 2
#define SEQ 2048
#define CDIM 2048
#define NHEAD 16
#define NKVHEAD 4
#define HDIM 128
#define WINDOW 512
#define QKV_N ((NHEAD + 2 * NKVHEAD) * HDIM) // 3072

typedef __attribute__((ext_vector_type(16))) __bf16 v16bf;
typedef __attribute__((ext_vector_type(8)))  float  v8f;
typedef unsigned int u32x4 __attribute__((ext_vector_type(4)));
typedef int          i32x8 __attribute__((ext_vector_type(8)));
typedef int          i32x4 __attribute__((ext_vector_type(4)));

union FragB16 { v16bf v; unsigned int u[8]; };

static __device__ __forceinline__ unsigned short f32_to_bf16(float f) {
    unsigned int u = __float_as_uint(f);
    u += 0x7FFFu + ((u >> 16) & 1u);     // round-to-nearest-even
    return (unsigned short)(u >> 16);
}
static __device__ __forceinline__ float bf16_to_f32(unsigned short h) {
    return __uint_as_float(((unsigned int)h) << 16);
}
static __device__ __forceinline__ v8f vzero8() {
    v8f z;
#pragma unroll
    for (int i = 0; i < 8; ++i) z[i] = 0.0f;
    return z;
}
// reduction over a 16-lane half-wave (xor masks < 16 stay inside the half)
static __device__ __forceinline__ float hmax16(float v) {
#pragma unroll
    for (int m = 1; m < 16; m <<= 1) v = fmaxf(v, __shfl_xor(v, m, 32));
    return v;
}
static __device__ __forceinline__ float hsum16(float v) {
#pragma unroll
    for (int m = 1; m < 16; m <<= 1) v += __shfl_xor(v, m, 32);
    return v;
}
// dword index (k0/2) of bf16 A/B fragment pack v for this half-wave
static __device__ __forceinline__ int frag_kdw(int v, int half) {
    return v + (v >= 4 ? 4 : 0) + (half ? 4 : 0);
}
// wave-relative LDS byte offset (generic AS0 ptr low 32 bits == LDS offset,
// per ISA 10.2 aperture mapping)
static __device__ __forceinline__ unsigned lds_off32(const void* p) {
    return (unsigned)(unsigned long long)(size_t)p;
}

// ---------------------------------------------------------------------------
// fp32 -> bf16 conversion
// ---------------------------------------------------------------------------
__global__ void cvt_f32_to_bf16(const float* __restrict__ in,
                                unsigned short* __restrict__ out, int n) {
    int i = blockIdx.x * blockDim.x + threadIdx.x;
    int stride = gridDim.x * blockDim.x;
    for (; i < n; i += stride) out[i] = f32_to_bf16(in[i]);
}

// ---------------------------------------------------------------------------
// Tiled bf16 WMMA GEMM: C(MxN) = A(MxK) * B(KxN), fp32 accumulate.
// Block tile 128x64, BK=32. 256 threads = 8 waves.
//  - A tile staged via GLOBAL_LOAD_ASYNC_TO_LDS_B128 (ASYNCcnt)
//  - B tile staged via Tensor Data Mover (TENSORcnt), wave 0 issues one D#
// ---------------------------------------------------------------------------
template <bool OUT_BF16>
__global__ __launch_bounds__(256) void gemm_bf16_wmma(
    const unsigned short* __restrict__ A,   // M x K bf16
    const unsigned short* __restrict__ Bm,  // K x N bf16
    void* __restrict__ Out, int M, int N, int K) {
    __shared__ unsigned int AsU[128 * 16];  // 128 rows x 32 k (bf16)
    __shared__ unsigned int BsU[32 * 32];   // 32 k x 64 n (bf16), row-major

    const int tid  = threadIdx.x;
    const int wave = tid >> 5;
    const int lane = tid & 31;
    const int half = lane >> 4;
    const int rowm = lane & 15;
    const int row0 = blockIdx.y * 128;
    const int col0 = blockIdx.x * 64;

    const unsigned AsOff = lds_off32(AsU);
    const unsigned BsOff = lds_off32(BsU);
    const unsigned short* Bs16 = (const unsigned short*)BsU;

    v8f acc[4];
#pragma unroll
    for (int nt = 0; nt < 4; ++nt) acc[nt] = vzero8();

    const int kTiles = K >> 5;
    for (int kt = 0; kt < kTiles; ++kt) {
        const int k0 = kt << 5;

        // ---- A tile: 512 async b128 copies (2 per thread) ----
#pragma unroll
        for (int i = 0; i < 2; ++i) {
            int idx = tid + i * 256;            // b128 chunk id
            int r = idx >> 2, c = idx & 3;      // 4 x 16B per 32-elem row
            const unsigned short* gp = A + (size_t)(row0 + r) * K + k0 + c * 8;
            unsigned ldst = AsOff + idx * 16;
            asm volatile("global_load_async_to_lds_b128 %0, %1, off"
                         :: "v"(ldst), "v"((unsigned long long)(size_t)gp)
                         : "memory");
        }

        // ---- B tile (32k x 64n) via TDM, one descriptor from wave 0 ----
        if (wave == 0) {
            unsigned long long gB =
                (unsigned long long)(size_t)(Bm + (size_t)k0 * N + col0);
            u32x4 g0;
            g0[0] = 1u;                                   // count=1, no gather
            g0[1] = BsOff;                                // lds_addr
            g0[2] = (unsigned)gB;                         // global_addr[31:0]
            g0[3] = (unsigned)((gB >> 32) & 0x1FFFFFFu)   // global_addr[56:32]
                    | (2u << 30);                         // type=2 (image)
            i32x8 g1;
            g1[0] = 0x00010000;                           // data_size=2 bytes
            g1[1] = (int)((unsigned)(N & 0xffff) << 16);  // tensor_dim0 lo16
            g1[2] = (int)(((unsigned)(N >> 16) & 0xffffu) |
                          ((unsigned)(K & 0xffff) << 16));// dim0 hi / dim1 lo
            g1[3] = (int)(((unsigned)(K >> 16) & 0xffffu) |
                          (64u << 16));                   // dim1 hi / tile_dim0
            g1[4] = 32;                                   // tile_dim1 (tile_dim2=0)
            g1[5] = (int)N;                               // tensor_dim0_stride lo32
            g1[6] = 0;
            g1[7] = 0;
            i32x4 gz4 = {0, 0, 0, 0};
            i32x8 gz8 = {0, 0, 0, 0, 0, 0, 0, 0};
            __builtin_amdgcn_tensor_load_to_lds(g0, g1, gz4, gz4, gz8, 0);
            __builtin_amdgcn_s_wait_tensorcnt((short)0);
        }
        asm volatile("s_wait_asynccnt 0" ::: "memory");
        __syncthreads();

        if (kt + 1 < kTiles) {  // prefetch next K tile -> global_prefetch_b8
            __builtin_prefetch(A + (size_t)(row0 + (tid & 127)) * K + k0 + 32, 0, 1);
            __builtin_prefetch(Bm + (size_t)(k0 + 32 + (tid & 31)) * N + col0 +
                               ((tid >> 5) & 1) * 32, 0, 1);
        }

        // ---- build fragments & multiply ----
        FragB16 af;
        const int arow = wave * 16 + rowm;
#pragma unroll
        for (int v = 0; v < 8; ++v) af.u[v] = AsU[arow * 16 + frag_kdw(v, half)];
#pragma unroll
        for (int nt = 0; nt < 4; ++nt) {
            FragB16 bf;
            const int ncol = nt * 16 + rowm;
#pragma unroll
            for (int v = 0; v < 8; ++v) {
                int ke = 2 * frag_kdw(v, half);
                unsigned lo = Bs16[ke * 64 + ncol];
                unsigned hi = Bs16[(ke + 1) * 64 + ncol];
                bf.u[v] = lo | (hi << 16);
            }
            acc[nt] = __builtin_amdgcn_wmma_f32_16x16x32_bf16(
                false, af.v, false, bf.v, (short)0, acc[nt], false, false);
        }
        __syncthreads();
    }

    // ---- epilogue: C/D layout lane->(M = i + 8*half, N = lane&15) ----
#pragma unroll
    for (int nt = 0; nt < 4; ++nt) {
        int col = col0 + nt * 16 + rowm;
#pragma unroll
        for (int i = 0; i < 8; ++i) {
            int row = row0 + wave * 16 + i + half * 8;
            float val = acc[nt][i];
            if (OUT_BF16)
                ((unsigned short*)Out)[(size_t)row * N + col] = f32_to_bf16(val);
            else
                ((float*)Out)[(size_t)row * N + col] = val;
        }
    }
}

// ---------------------------------------------------------------------------
// RoPE + head split: qkv(B,T,3072) bf16 -> Q(B,16,T,128), K(B,4,T,128),
// V(B,4,T,128) bf16. freqs: (T, 64, 2) fp32.
// ---------------------------------------------------------------------------
__global__ __launch_bounds__(256) void rope_split_kernel(
    const unsigned short* __restrict__ qkv, const float* __restrict__ freqs,
    unsigned short* __restrict__ Qb, unsigned short* __restrict__ Kb,
    unsigned short* __restrict__ Vb) {
    const int t = blockIdx.x;
    const int b = blockIdx.y;
    const unsigned short* src = qkv + ((size_t)b * SEQ + t) * QKV_N;
    for (int p = threadIdx.x; p < QKV_N / 2; p += 256) {
        int c0 = p * 2;
        float x0 = bf16_to_f32(src[c0]);
        float x1 = bf16_to_f32(src[c0 + 1]);
        if (c0 < NHEAD * HDIM) {                       // Q
            int h = c0 >> 7, d = c0 & 127, dp = d >> 1;
            float cs = freqs[((size_t)t * 64 + dp) * 2 + 0];
            float sn = freqs[((size_t)t * 64 + dp) * 2 + 1];
            size_t o = (((size_t)b * NHEAD + h) * SEQ + t) * HDIM + d;
            Qb[o]     = f32_to_bf16(x0 * cs - x1 * sn);
            Qb[o + 1] = f32_to_bf16(x0 * sn + x1 * cs);
        } else if (c0 < (NHEAD + NKVHEAD) * HDIM) {    // K
            int cc = c0 - NHEAD * HDIM;
            int h = cc >> 7, d = cc & 127, dp = d >> 1;
            float cs = freqs[((size_t)t * 64 + dp) * 2 + 0];
            float sn = freqs[((size_t)t * 64 + dp) * 2 + 1];
            size_t o = (((size_t)b * NKVHEAD + h) * SEQ + t) * HDIM + d;
            Kb[o]     = f32_to_bf16(x0 * cs - x1 * sn);
            Kb[o + 1] = f32_to_bf16(x0 * sn + x1 * cs);
        } else {                                        // V passthrough
            int cc = c0 - (NHEAD + NKVHEAD) * HDIM;
            int h = cc >> 7, d = cc & 127;
            size_t o = (((size_t)b * NKVHEAD + h) * SEQ + t) * HDIM + d;
            Vb[o] = src[c0]; Vb[o + 1] = src[c0 + 1];
        }
    }
}

// ---------------------------------------------------------------------------
// Sliding-window flash attention, 1 wave per (b, h, 16-query tile).
// ---------------------------------------------------------------------------
__global__ __launch_bounds__(32) void swa_attention_kernel(
    const unsigned short* __restrict__ Qb, const unsigned short* __restrict__ Kb,
    const unsigned short* __restrict__ Vb, unsigned short* __restrict__ Yb) {
    __shared__ unsigned short Ps[16 * 32];   // P tile, [row][k] bf16
    __shared__ unsigned short Vt[128 * 32];  // V transposed, [d][k] bf16

    const int lane = threadIdx.x;
    const int half = lane >> 4;
    const int rowm = lane & 15;
    const int blk  = blockIdx.x;
    const int tile = blk & (SEQ / 16 - 1);
    const int h    = (blk >> 7) & (NHEAD - 1);
    const int b    = blk >> 11;
    const int q0   = tile * 16;
    const int kvh  = h >> 2;                 // GQA: 4 Q heads share a KV head

    const unsigned short* Q  = Qb + ((size_t)b * NHEAD + h) * SEQ * HDIM;
    const unsigned short* Kp = Kb + ((size_t)b * NKVHEAD + kvh) * SEQ * HDIM;
    const unsigned short* Vp = Vb + ((size_t)b * NKVHEAD + kvh) * SEQ * HDIM;

    // Q fragments: 4 chunks of 16x32 along D=128 (A-matrix layout)
    FragB16 qf[4];
#pragma unroll
    for (int c = 0; c < 4; ++c)
#pragma unroll
        for (int v = 0; v < 8; ++v) {
            int d = c * 32 + 2 * frag_kdw(v, half);
            qf[c].u[v] = *(const unsigned int*)(Q + (size_t)(q0 + rowm) * HDIM + d);
        }

    float mrow[8], lrow[8];
    v8f oacc[8];
#pragma unroll
    for (int i = 0; i < 8; ++i) { mrow[i] = -3.0e38f; lrow[i] = 0.0f; }
#pragma unroll
    for (int n = 0; n < 8; ++n) oacc[n] = vzero8();

    const float scale = 0.08838834764831845f;  // 1/sqrt(128)
    int kmin = q0 - (WINDOW - 1); if (kmin < 0) kmin = 0;
    const int kt0 = kmin >> 4;
    const int nkt = tile - kt0 + 1;

    for (int it = 0; it < nkt; it += 2) {
        const int ktA = kt0 + it;
        const int ktB = ktA + 1;             // may be out of range -> masked
        v8f s0 = vzero8(), s1 = vzero8();
        int kkA = ktA * 16 + rowm; if (kkA > SEQ - 1) kkA = SEQ - 1;
        int kkB = ktB * 16 + rowm; if (kkB > SEQ - 1) kkB = SEQ - 1;
#pragma unroll
        for (int c = 0; c < 4; ++c) {
            FragB16 kf;
#pragma unroll
            for (int v = 0; v < 8; ++v)
                kf.u[v] = *(const unsigned int*)(Kp + (size_t)kkA * HDIM +
                                                 c * 32 + 2 * frag_kdw(v, half));
            s0 = __builtin_amdgcn_wmma_f32_16x16x32_bf16(
                false, qf[c].v, false, kf.v, (short)0, s0, false, false);
        }
#pragma unroll
        for (int c = 0; c < 4; ++c) {
            FragB16 kf;
#pragma unroll
            for (int v = 0; v < 8; ++v)
                kf.u[v] = *(const unsigned int*)(Kp + (size_t)kkB * HDIM +
                                                 c * 32 + 2 * frag_kdw(v, half));
            s1 = __builtin_amdgcn_wmma_f32_16x16x32_bf16(
                false, qf[c].v, false, kf.v, (short)0, s1, false, false);
        }
        // ---- mask + online softmax update (per-row, in-lane) ----
        float p0[8], p1[8], alpha[8];
#pragma unroll
        for (int i = 0; i < 8; ++i) {
            int q = q0 + i + half * 8;
            int keyA = ktA * 16 + rowm;
            int keyB = ktB * 16 + rowm;
            float a  = ((keyA <= q) && (q - keyA < WINDOW)) ? s0[i] * scale : -3.0e38f;
            float bb = ((keyB <= q) && (q - keyB < WINDOW)) ? s1[i] * scale : -3.0e38f;
            float mt = hmax16(fmaxf(a, bb));
            float mnew = fmaxf(mrow[i], mt);
            float e0 = __expf(a - mnew);
            float e1 = __expf(bb - mnew);
            alpha[i] = __expf(mrow[i] - mnew);
            mrow[i] = mnew;
            lrow[i] = lrow[i] * alpha[i] + hsum16(e0 + e1);
            p0[i] = e0; p1[i] = e1;
        }
#pragma unroll
        for (int n = 0; n < 8; ++n)
#pragma unroll
            for (int i = 0; i < 8; ++i) oacc[n][i] *= alpha[i];

        // ---- P (C/D layout) -> LDS -> A-fragment (16x32 bf16) ----
#pragma unroll
        for (int i = 0; i < 8; ++i) {
            int r = i + half * 8;
            Ps[r * 32 + rowm]      = f32_to_bf16(p0[i]);
            Ps[r * 32 + 16 + rowm] = f32_to_bf16(p1[i]);
        }
        // ---- stage V (32 keys x 128 d) transposed into Vt[d][k] ----
#pragma unroll 4
        for (int j = 0; j < 32; ++j) {
            int key = ktA * 16 + j; if (key > SEQ - 1) key = SEQ - 1;
            const unsigned int* vr = (const unsigned int*)(Vp + (size_t)key * HDIM);
            unsigned int d0 = vr[lane];
            unsigned int d1 = vr[lane + 32];
            Vt[(2 * lane     ) * 32 + j] = (unsigned short)(d0 & 0xffffu);
            Vt[(2 * lane + 1 ) * 32 + j] = (unsigned short)(d0 >> 16);
            Vt[(2 * lane + 64) * 32 + j] = (unsigned short)(d1 & 0xffffu);
            Vt[(2 * lane + 65) * 32 + j] = (unsigned short)(d1 >> 16);
        }
        asm volatile("s_wait_dscnt 0" ::: "memory");

        FragB16 pf;
#pragma unroll
        for (int v = 0; v < 8; ++v)
            pf.u[v] = ((const unsigned int*)Ps)[rowm * 16 + frag_kdw(v, half)];
#pragma unroll
        for (int n = 0; n < 8; ++n) {
            FragB16 vf;
            const int dcol = n * 16 + rowm;
#pragma unroll
            for (int v = 0; v < 8; ++v)
                vf.u[v] = ((const unsigned int*)Vt)[dcol * 16 + frag_kdw(v, half)];
            oacc[n] = __builtin_amdgcn_wmma_f32_16x16x32_bf16(
                false, pf.v, false, vf.v, (short)0, oacc[n], false, false);
        }
    }

    // ---- epilogue: y(B,T,C) bf16, col = h*128 + d ----
#pragma unroll
    for (int n = 0; n < 8; ++n) {
        int d = n * 16 + rowm;
#pragma unroll
        for (int i = 0; i < 8; ++i) {
            int q = q0 + i + half * 8;
            float val = oacc[n][i] / lrow[i];
            Yb[((size_t)b * SEQ + q) * CDIM + h * HDIM + d] = f32_to_bf16(val);
        }
    }
}

// ---------------------------------------------------------------------------
// Host-side orchestration
// ---------------------------------------------------------------------------
extern "C" void kernel_launch(void* const* d_in, const int* in_sizes, int n_in,
                              void* d_out, int out_size, void* d_ws, size_t ws_size,
                              hipStream_t stream) {
    (void)in_sizes; (void)n_in; (void)out_size; (void)ws_size;
    const float* x      = (const float*)d_in[0];  // (B,T,C)
    const float* w_attn = (const float*)d_in[1];  // (C, 3072)
    const float* w_proj = (const float*)d_in[2];  // (C, C)
    const float* freqs  = (const float*)d_in[3];  // (T, 64, 2)
    // d_in[4] = mask (all ones); window logic reproduces mask & ~in_window.

    const int M = BATCH * SEQ;  // 4096
    size_t off = 0;
    auto alloc = [&](size_t bytes) {
        void* p = (char*)d_ws + off;
        off += (bytes + 255) & ~(size_t)255;
        return p;
    };
    unsigned short* xb   = (unsigned short*)alloc((size_t)M * CDIM * 2);
    unsigned short* wab  = (unsigned short*)alloc((size_t)CDIM * QKV_N * 2);
    unsigned short* wpb  = (unsigned short*)alloc((size_t)CDIM * CDIM * 2);
    unsigned short* qkvb = (unsigned short*)alloc((size_t)M * QKV_N * 2);
    unsigned short* Qbuf = (unsigned short*)alloc((size_t)BATCH * NHEAD * SEQ * HDIM * 2);
    unsigned short* Kbuf = (unsigned short*)alloc((size_t)BATCH * NKVHEAD * SEQ * HDIM * 2);
    unsigned short* Vbuf = (unsigned short*)alloc((size_t)BATCH * NKVHEAD * SEQ * HDIM * 2);
    unsigned short* yb   = (unsigned short*)alloc((size_t)M * CDIM * 2);

    cvt_f32_to_bf16<<<4096, 256, 0, stream>>>(x, xb, M * CDIM);
    cvt_f32_to_bf16<<<4096, 256, 0, stream>>>(w_attn, wab, CDIM * QKV_N);
    cvt_f32_to_bf16<<<4096, 256, 0, stream>>>(w_proj, wpb, CDIM * CDIM);

    gemm_bf16_wmma<true><<<dim3(QKV_N / 64, M / 128), 256, 0, stream>>>(
        xb, wab, qkvb, M, QKV_N, CDIM);

    rope_split_kernel<<<dim3(SEQ, BATCH), 256, 0, stream>>>(
        qkvb, freqs, Qbuf, Kbuf, Vbuf);

    swa_attention_kernel<<<BATCH * NHEAD * (SEQ / 16), 32, 0, stream>>>(
        Qbuf, Kbuf, Vbuf, yb);

    gemm_bf16_wmma<false><<<dim3(CDIM / 64, M / 128), 256, 0, stream>>>(
        yb, wpb, d_out, M, CDIM, CDIM);
}
